// MoELanguageZone_29480655520333
// MI455X (gfx1250) — matrix operand order repaired and checked by reading, not compiled, not run on hardware
//
#include <hip/hip_runtime.h>
#include <hip/hip_bf16.h>
#include <math.h>
#include <stdint.h>

// ---- model dims ----
#define VOCAB 32000
#define EDIM  512
#define HDIM  1024
#define MDIM  64
#define NEXP  8
#define HHDIM 512
#define BB    2
#define SS    2048
#define NTOK  (BB*SS)   // 4096

typedef __attribute__((ext_vector_type(16))) __bf16 v16bf;
typedef __attribute__((ext_vector_type(8)))  float  v8f;
typedef int v4i32 __attribute__((vector_size(16)));   // matches builtin param type

union F16x16 { v16bf v; uint4 u[2]; };

// ---- gfx1250 async global->LDS path (probe via __has_builtin) ----
#if defined(__HIP_DEVICE_COMPILE__) && __has_builtin(__builtin_amdgcn_global_load_async_to_lds_b128)
#define USE_ASYNC_LDS 1
#else
#define USE_ASYNC_LDS 0
#endif

__device__ __forceinline__ void async_wait_all() {
#if USE_ASYNC_LDS
#if __has_builtin(__builtin_amdgcn_s_wait_asynccnt)
  __builtin_amdgcn_s_wait_asynccnt(0);
#else
  asm volatile("s_wait_asynccnt 0" ::: "memory");
#endif
#endif
}

__device__ __forceinline__ void copy16_to_lds(const __bf16* g, __bf16* l) {
#if USE_ASYNC_LDS
  // builtin wants (AS1 v4i32*, AS3 v4i32*, imm offset, imm cpol);
  // generic LDS address low 32 bits == LDS byte offset (aperture layout)
  __builtin_amdgcn_global_load_async_to_lds_b128(
      (__attribute__((address_space(1))) v4i32*)(uintptr_t)g,
      (__attribute__((address_space(3))) v4i32*)(uintptr_t)l, 0, 0);
#else
  *(uint4*)l = *(const uint4*)g;
#endif
}

// ---- gfx1250 hardware tanh (V_TANH_F32 trans op) with branch-free fallback
__device__ __forceinline__ float fast_tanh(float x) {
#if defined(__HIP_DEVICE_COMPILE__) && __has_builtin(__builtin_amdgcn_tanhf)
  return __builtin_amdgcn_tanhf(x);
#elif defined(__HIP_DEVICE_COMPILE__) && __has_builtin(__builtin_amdgcn_tanh_f32)
  return __builtin_amdgcn_tanh_f32(x);
#else
  float t = __expf(2.f * x);
  return 1.f - 2.f / (t + 1.f);
#endif
}
// sigmoid(z) == 0.5*(1 + tanh(z/2)) exactly
__device__ __forceinline__ float fast_sigmoid(float z) {
  return 0.5f * fast_tanh(0.5f * z) + 0.5f;
}

// A-fragment (16x32 bf16, M x K): lane<16 -> row=lane, K {0..7,16..23};
// lane>=16 -> row=lane-16, K {8..15,24..31}  (ISA 7.12.2)
__device__ __forceinline__ v16bf frag_a(const __bf16* base, int ld, int lane) {
  int row = lane & 15;
  int hi  = (lane >> 4) << 4;               // 0 or 16 bytes
  const char* p = (const char*)(base + (size_t)row * ld);
  F16x16 f;
  f.u[0] = *(const uint4*)(p + hi);
  f.u[1] = *(const uint4*)(p + 32 + hi);
  return f.v;
}
// B-fragment (32x16 bf16, K x N) from K-major (transposed) storage:
// lane<16 -> col=lane, K {0..7,16..23}; lane>=16 -> K {8..15,24..31}
__device__ __forceinline__ v16bf frag_b(const __bf16* baseT, int ld, int lane) {
  int col = lane & 15;
  int hi  = (lane >> 4) << 4;               // 0 or 16 bytes
  const char* p = (const char*)(baseT + (size_t)col * ld) + hi;
  F16x16 f;
  f.u[0] = *(const uint4*)(p);
  f.u[1] = *(const uint4*)(p + 32);
  return f.v;
}

// ---- fp32 [R][C] -> bf16 transposed [C][R], tiled through LDS ----
__global__ void transpose_cvt_bf16(const float* __restrict__ in,
                                   __bf16* __restrict__ out, int R, int C) {
  __shared__ float tile[32][33];
  size_t bo = (size_t)blockIdx.z * R * C;
  in += bo; out += bo;
  int c0 = blockIdx.x * 32, r0 = blockIdx.y * 32;
  for (int i = threadIdx.y; i < 32; i += 8) {
    int r = r0 + i, c = c0 + threadIdx.x;
    tile[i][threadIdx.x] = (r < R && c < C) ? in[(size_t)r * C + c] : 0.f;
  }
  __syncthreads();
  for (int i = threadIdx.y; i < 32; i += 8) {
    int c = c0 + i, r = r0 + threadIdx.x;
    if (c < C && r < R) out[(size_t)c * R + r] = (__bf16)tile[threadIdx.x][i];
  }
}

// ---- embedding gather -> bf16 ----
__global__ void embed_gather(const int* __restrict__ ids,
                             const float* __restrict__ emb,
                             __bf16* __restrict__ x) {
  int idx = blockIdx.x * blockDim.x + threadIdx.x;   // NTOK*EDIM
  int t = idx >> 9, e = idx & 511;                   // EDIM=512
  x[idx] = (__bf16)emb[(size_t)ids[t] * EDIM + e];
}

// ---- GIF spiking scan: sequential over S, parallel over (b,h) chains ----
__global__ void gif_scan(const float* __restrict__ u, __bf16* __restrict__ sp,
                         int Hd) {
  int idx = blockIdx.x * blockDim.x + threadIdx.x;   // BB*Hd chains
  int b = idx / Hd, h = idx - b * Hd;
  const float* up = u  + (size_t)b * SS * Hd + h;
  __bf16*      op = sp + (size_t)b * SS * Hd + h;
  float v = 0.f;
  for (int s = 0; s < SS; ++s) {
    if ((s & 7) == 0 && s + 16 < SS)
      __builtin_prefetch(up + (size_t)(s + 16) * Hd, 0, 0);  // global_prefetch_b8
    v = 0.9f * v + up[(size_t)s * Hd];
    float sg = fast_sigmoid(4.f * (v - 1.f));         // hw v_tanh when available
    op[(size_t)s * Hd] = (__bf16)sg;
    v -= sg;                                          // soft reset, thresh=1
  }
}

// ---- bf16 WMMA GEMM: C[M,N] = A[M,K] @ Bt[N,K]^T (+bias), fp32 acc ----
// Templated tile: BM=WGM*TM*16, BN=WGN*TN*16, BK=64, double-buffered LDS,
// async global->LDS staging on gfx1250 when available.
template<int WGM, int WGN, int TM, int TN>
__global__ void __launch_bounds__(WGM * WGN * 32)
gemm_bf16(const __bf16* __restrict__ A, const __bf16* __restrict__ Bt,
          float* __restrict__ C, __bf16* __restrict__ Cbf,
          const float* __restrict__ bias, int Nc, int K) {
  constexpr int BM = WGM * TM * 16;
  constexpr int BN = WGN * TN * 16;
  constexpr int BK = 64;
  constexpr int LSTR = BK + 8;                 // 72 elems = 144B, 16B aligned
  constexpr int NTHR = WGM * WGN * 32;
  __shared__ __bf16 As[2][BM * LSTR];
  __shared__ __bf16 Bs[2][BN * LSTR];
  int tid = threadIdx.x, lane = tid & 31, wave = tid >> 5;
  int m0 = blockIdx.x * BM, n0 = blockIdx.y * BN;
  int wm = (wave % WGM) * (TM * 16);
  int wn = (wave / WGM) * (TN * 16);
  v8f acc[TM][TN] = {};

  auto stage = [&](int buf, int k0) {
    constexpr int CHA = BM * BK / 16;          // 16-elem (32B) chunks
    constexpr int CHB = BN * BK / 16;
    for (int c = tid; c < CHA; c += NTHR) {
      int row = c / (BK / 16), off = (c % (BK / 16)) * 16;
      copy16_to_lds(A + (size_t)(m0 + row) * K + k0 + off,
                    &As[buf][row * LSTR + off]);
    }
    for (int c = tid; c < CHB; c += NTHR) {
      int row = c / (BK / 16), off = (c % (BK / 16)) * 16;
      copy16_to_lds(Bt + (size_t)(n0 + row) * K + k0 + off,
                    &Bs[buf][row * LSTR + off]);
    }
  };

  stage(0, 0);
  async_wait_all();
  __syncthreads();
  int buf = 0;
  for (int k0 = 0; k0 < K; k0 += BK) {
    if (k0 + BK < K) stage(buf ^ 1, k0 + BK);  // prefetch next tile (async)
    for (int ks = 0; ks < BK / 32; ++ks) {
      v16bf fa[TM], fb[TN];
      for (int i = 0; i < TM; ++i)
        fa[i] = frag_a(&As[buf][(wm + i * 16) * LSTR + ks * 32], LSTR, lane);
      for (int j = 0; j < TN; ++j)
        fb[j] = frag_b(&Bs[buf][(wn + j * 16) * LSTR + ks * 32], LSTR, lane);
      for (int i = 0; i < TM; ++i)
        for (int j = 0; j < TN; ++j)
          acc[i][j] = __builtin_amdgcn_wmma_f32_16x16x32_bf16(
              false, fa[i], false, fb[j], (short)0, acc[i][j], false, false);
    }
    async_wait_all();
    __syncthreads();
    buf ^= 1;
  }

  int rbase = m0 + wm + ((lane >> 4) << 3);    // +8 for upper half-wave
  int cbase = n0 + wn + (lane & 15);
  for (int i = 0; i < TM; ++i)
    for (int j = 0; j < TN; ++j) {
      int col = cbase + j * 16;
      float bv = bias ? bias[col] : 0.f;
      for (int q = 0; q < 8; ++q) {
        int row = rbase + i * 16 + q;
        float val = acc[i][j][q] + bv;
        C[(size_t)row * Nc + col] = val;
        if (Cbf) Cbf[(size_t)row * Nc + col] = (__bf16)val;
      }
    }
}

// ---- router: per-token 64->64 tanh MLP -> 8 logits -> softmax -> top2 ----
__global__ void router_kernel(const float* __restrict__ cont,
                              const float* __restrict__ rW1, const float* __restrict__ rb1,
                              const float* __restrict__ rW2, const float* __restrict__ rb2,
                              float* __restrict__ gsel, int* __restrict__ isel) {
  int t = blockIdx.x * blockDim.x + threadIdx.x;
  float c[64];
  const float* cp = cont + (size_t)t * MDIM;
  for (int m = 0; m < 64; ++m) c[m] = cp[m];
  float gl[8];
  for (int e = 0; e < 8; ++e) gl[e] = rb2[e];
  for (int j = 0; j < 64; ++j) {
    float hs = rb1[j];
    for (int m = 0; m < 64; ++m) hs += c[m] * rW1[m * 64 + j];
    float tj = fast_tanh(hs);
    for (int e = 0; e < 8; ++e) gl[e] += tj * rW2[j * 8 + e];
  }
  float mx = gl[0];
  for (int e = 1; e < 8; ++e) mx = fmaxf(mx, gl[e]);
  float p[8];
  for (int e = 0; e < 8; ++e) p[e] = __expf(gl[e] - mx);
  int i0 = 0; float v0 = p[0];
  for (int e = 1; e < 8; ++e) if (p[e] > v0) { v0 = p[e]; i0 = e; }
  int i1 = (i0 == 0) ? 1 : 0; float v1 = p[i1];
  for (int e = 0; e < 8; ++e) if (e != i0 && p[e] > v1) { v1 = p[e]; i1 = e; }
  float inv = 1.f / (v0 + v1);                 // top-2 ratio: softmax denom cancels
  gsel[t * 2 + 0] = v0 * inv; gsel[t * 2 + 1] = v1 * inv;
  isel[t * 2 + 0] = i0;       isel[t * 2 + 1] = i1;
}

// ---- experts: eo[t,e] = tanh(cont@eW1[e]+eb1[e]) . eW2[e] + eb2[e]
// fused WMMA + tanh + dot + cross-lane reduce; h never materialized.
__global__ void __launch_bounds__(128)
experts_kernel(const __bf16* __restrict__ contb, const __bf16* __restrict__ eW1t,
               const float* __restrict__ eb1, const float* __restrict__ eW2,
               const float* __restrict__ eb2, float* __restrict__ eo) {
  int t0 = blockIdx.x * 16, e = blockIdx.y;
  int tid = threadIdx.x, lane = tid & 31, wave = tid >> 5;
  __shared__ float red[4][16];
  v16bf a0 = frag_a(contb + (size_t)t0 * MDIM, MDIM, lane);       // K 0..31
  v16bf a1 = frag_a(contb + (size_t)t0 * MDIM + 32, MDIM, lane);  // K 32..63
  const __bf16* w = eW1t + (size_t)e * HHDIM * MDIM;              // [HH][M]
  float ps[8] = {0,0,0,0,0,0,0,0};
  for (int jn = wave; jn < HHDIM / 16; jn += 4) {
    int n0 = jn * 16;
    v16bf b0 = frag_b(w + (size_t)n0 * MDIM, MDIM, lane);
    v16bf b1 = frag_b(w + (size_t)n0 * MDIM + 32, MDIM, lane);
    v8f cacc = {};
    cacc = __builtin_amdgcn_wmma_f32_16x16x32_bf16(false, a0, false, b0, (short)0, cacc, false, false);
    cacc = __builtin_amdgcn_wmma_f32_16x16x32_bf16(false, a1, false, b1, (short)0, cacc, false, false);
    int n = n0 + (lane & 15);
    float bv = eb1[e * HHDIM + n];
    float w2 = eW2[e * HHDIM + n];
    for (int q = 0; q < 8; ++q) ps[q] += fast_tanh(cacc[q] + bv) * w2;
  }
  for (int m = 1; m <= 8; m <<= 1)
    for (int q = 0; q < 8; ++q) ps[q] += __shfl_xor(ps[q], m, 32);
  if ((lane & 15) == 0) {
    int off = (lane >> 4) * 8;                 // lane0: rows0-7, lane16: rows8-15
    for (int q = 0; q < 8; ++q) red[wave][off + q] = ps[q];
  }
  __syncthreads();
  if (tid < 16) {
    float s = red[0][tid] + red[1][tid] + red[2][tid] + red[3][tid] + eb2[e];
    eo[(size_t)(t0 + tid) * NEXP + e] = s;
  }
}

__global__ void combine_kernel(const float* __restrict__ eo,
                               const float* __restrict__ gsel,
                               const int* __restrict__ isel,
                               float* __restrict__ y) {
  int t = blockIdx.x * blockDim.x + threadIdx.x;
  y[t] = gsel[t * 2] * eo[(size_t)t * NEXP + isel[t * 2]]
       + gsel[t * 2 + 1] * eo[(size_t)t * NEXP + isel[t * 2 + 1]];
}

// colsum of c2s_W: y_rep @ c2s_W == y * colsum (all M cols of y_rep equal y)
__global__ void csum_kernel(const float* __restrict__ c2sW, float* __restrict__ cs) {
  int h = blockIdx.x * blockDim.x + threadIdx.x;
  float s = 0.f;
  for (int m = 0; m < MDIM; ++m) s += c2sW[(size_t)m * HDIM + h];
  cs[h] = s;
}

__global__ void rate_kernel(const float* __restrict__ y, const float* __restrict__ cs,
                            const float* __restrict__ c2sb, __bf16* __restrict__ rate) {
  int idx = blockIdx.x * blockDim.x + threadIdx.x;   // NTOK*HDIM
  int t = idx >> 10, h = idx & 1023;                 // HDIM=1024
  float r = fast_sigmoid(y[t] * cs[h] + c2sb[h]);
  rate[idx] = (__bf16)r;
}

extern "C" void kernel_launch(void* const* d_in, const int* in_sizes, int n_in,
                              void* d_out, int out_size, void* d_ws, size_t ws_size,
                              hipStream_t stream) {
  const int*   ids   = (const int*)  d_in[0];
  const float* emb   = (const float*)d_in[1];
  const float* encW  = (const float*)d_in[2];
  const float* encb  = (const float*)d_in[3];
  const float* s2cW  = (const float*)d_in[4];
  const float* s2cb  = (const float*)d_in[5];
  const float* rW1   = (const float*)d_in[6];
  const float* rb1   = (const float*)d_in[7];
  const float* rW2   = (const float*)d_in[8];
  const float* rb2   = (const float*)d_in[9];
  const float* eW1   = (const float*)d_in[10];
  const float* eb1   = (const float*)d_in[11];
  const float* eW2   = (const float*)d_in[12];
  const float* eb2   = (const float*)d_in[13];
  const float* c2sW  = (const float*)d_in[14];
  const float* c2sb  = (const float*)d_in[15];
  const float* decW  = (const float*)d_in[16];
  const float* decb  = (const float*)d_in[17];
  const float* outW  = (const float*)d_in[18];
  const float* outb  = (const float*)d_in[19];
  float* out = (float*)d_out;

  char* ws = (char*)d_ws;
  size_t off = 0;
  auto alloc = [&](size_t bytes) -> void* {
    void* p = ws + off;
    off = (off + bytes + 255) & ~(size_t)255;
    return p;
  };
  __bf16* xb     = (__bf16*)alloc((size_t)NTOK * EDIM * 2);
  __bf16* encWt  = (__bf16*)alloc((size_t)HDIM * EDIM * 2);
  float*  u1     = (float*) alloc((size_t)NTOK * HDIM * 4);
  __bf16* spb    = (__bf16*)alloc((size_t)NTOK * HDIM * 2);
  __bf16* s2cWt  = (__bf16*)alloc((size_t)MDIM * HDIM * 2);
  float*  cont   = (float*) alloc((size_t)NTOK * MDIM * 4);
  __bf16* contb  = (__bf16*)alloc((size_t)NTOK * MDIM * 2);
  __bf16* eW1t   = (__bf16*)alloc((size_t)NEXP * HHDIM * MDIM * 2);
  float*  eo     = (float*) alloc((size_t)NTOK * NEXP * 4);
  float*  gsel   = (float*) alloc((size_t)NTOK * 2 * 4);
  int*    isel   = (int*)   alloc((size_t)NTOK * 2 * 4);
  float*  cs     = (float*) alloc((size_t)HDIM * 4);
  float*  yv     = (float*) alloc((size_t)NTOK * 4);
  __bf16* rateb  = (__bf16*)alloc((size_t)NTOK * HDIM * 2);
  __bf16* decWt  = (__bf16*)alloc((size_t)EDIM * HDIM * 2);
  float*  u2     = (float*) alloc((size_t)NTOK * EDIM * 4);
  __bf16* decb_a = (__bf16*)alloc((size_t)NTOK * EDIM * 2);
  __bf16* outWt  = (__bf16*)alloc((size_t)VOCAB * EDIM * 2);

  dim3 tb(32, 8, 1);
  // weight transpose + bf16 convert
  transpose_cvt_bf16<<<dim3(HDIM/32, EDIM/32, 1),  tb, 0, stream>>>(encW, encWt, EDIM, HDIM);
  transpose_cvt_bf16<<<dim3(MDIM/32, HDIM/32, 1),  tb, 0, stream>>>(s2cW, s2cWt, HDIM, MDIM);
  transpose_cvt_bf16<<<dim3(HHDIM/32, MDIM/32, NEXP), tb, 0, stream>>>(eW1, eW1t, MDIM, HHDIM);
  transpose_cvt_bf16<<<dim3(EDIM/32, HDIM/32, 1),  tb, 0, stream>>>(decW, decWt, HDIM, EDIM);
  transpose_cvt_bf16<<<dim3(VOCAB/32, EDIM/32, 1), tb, 0, stream>>>(outW, outWt, EDIM, VOCAB);

  // embedding gather
  embed_gather<<<(NTOK * EDIM) / 256, 256, 0, stream>>>(ids, emb, xb);
  // encoder GEMM: u1 = x @ enc_W + enc_b   (4096x512x1024)  [128x128 tile]
  gemm_bf16<2,4,4,2><<<dim3(NTOK/128, HDIM/128), 256, 0, stream>>>(xb, encWt, u1, nullptr, encb, HDIM, EDIM);
  // GIF encoder scan
  gif_scan<<<(BB * HDIM) / 256, 256, 0, stream>>>(u1, spb, HDIM);
  // cont = sp @ s2c_W + s2c_b (fp32 + bf16 mirror)  (N=64 -> 64x64 tile)
  gemm_bf16<2,2,2,2><<<dim3(NTOK/64, MDIM/64), 128, 0, stream>>>(spb, s2cWt, cont, contb, s2cb, MDIM, HDIM);
  // router
  router_kernel<<<NTOK / 256, 256, 0, stream>>>(cont, rW1, rb1, rW2, rb2, gsel, isel);
  // experts (fused)
  experts_kernel<<<dim3(NTOK/16, NEXP), 128, 0, stream>>>(contb, eW1t, eb1, eW2, eb2, eo);
  // combine top-2
  combine_kernel<<<NTOK / 256, 256, 0, stream>>>(eo, gsel, isel, yv);
  // rate = sigmoid(y * colsum(c2s_W) + c2s_b)
  csum_kernel<<<HDIM / 256, 256, 0, stream>>>(c2sW, cs);
  rate_kernel<<<(NTOK * HDIM) / 256, 256, 0, stream>>>(yv, cs, c2sb, rateb);
  // decoder GEMM: u2 = rate @ dec_W + dec_b  (4096x1024x512)
  gemm_bf16<2,4,4,2><<<dim3(NTOK/128, EDIM/128), 256, 0, stream>>>(rateb, decWt, u2, nullptr, decb, EDIM, HDIM);
  // GIF decoder scan
  gif_scan<<<(BB * EDIM) / 256, 256, 0, stream>>>(u2, decb_a, EDIM);
  // logits GEMM: out = dec @ out_W + out_b   (134 GFLOP, the big one)
  gemm_bf16<2,4,4,2><<<dim3(NTOK/128, VOCAB/128), 256, 0, stream>>>(decb_a, outWt, out, nullptr, outb, VOCAB, EDIM);

  (void)in_sizes; (void)n_in; (void)out_size; (void)ws_size;
}
